// CustomDiceLoss_68848325755491
// MI455X (gfx1250) — compile-verified, matrix operand order checked by reading, not run on hardware
//
#include <hip/hip_runtime.h>
#include <hip/hip_bf16.h>

// Problem constants
#define BS 8
#define CH 512
#define HH 128
#define WW 128
#define HW (HH * WW)          // 16384 = 2^14
#define KCLS 512              // 2^9
#define NPIX (BS * HW)        // 131072

typedef __attribute__((ext_vector_type(8)))  float   f32x8;
typedef __attribute__((ext_vector_type(8)))  __bf16  bf16x8;
typedef __attribute__((ext_vector_type(16))) __bf16  bf16x16;

// ---------------- workspace layout (bytes) ----------------
#define WS_EMBB 0                                   // 512*512 bf16 = 524288
#define WS_RSB  (WS_EMBB + KCLS * CH * 2)           // 512 f32
#define WS_GTC  (WS_RSB + KCLS * 4)                 // 512 i32
#define WS_PRC  (WS_GTC + KCLS * 4)                 // 512 i32
#define WS_INT  (WS_PRC + KCLS * 4)                 // 512 i32
#define WS_GTL  (WS_INT + KCLS * 4)                 // NPIX i32
#define WS_PRL  (WS_GTL + NPIX * 4)                 // NPIX i32

// ---------------- kernel: embeddings f32 -> bf16 + rsqrt(norm^2) ----------------
__global__ __launch_bounds__(256) void dice_emb_prep(const float* __restrict__ emb,
                                                     __bf16* __restrict__ embb,
                                                     float* __restrict__ rsb) {
  const int k = blockIdx.x;
  const int t = threadIdx.x;
  const float* row = emb + (size_t)k * CH;
  float v0 = row[t];
  float v1 = row[t + 256];
  embb[(size_t)k * CH + t]        = (__bf16)v0;
  embb[(size_t)k * CH + t + 256]  = (__bf16)v1;
  float s = v0 * v0 + v1 * v1;
#pragma unroll
  for (int m = 16; m >= 1; m >>= 1) s += __shfl_xor(s, m, 32);
  __shared__ float wsum[8];
  if ((t & 31) == 0) wsum[t >> 5] = s;
  __syncthreads();
  if (t == 0) {
    float tot = 0.f;
#pragma unroll
    for (int i = 0; i < 8; ++i) tot += wsum[i];
    rsb[k] = rsqrtf(tot);
  }
}

// ---------------- kernel: zero the histogram counters ----------------
__global__ void dice_zero(int* __restrict__ cnt) {
  for (int i = threadIdx.x; i < 3 * KCLS; i += blockDim.x) cnt[i] = 0;
}

// ---------------- kernel: recover gt labels + gt histogram from one-hot ----------------
__global__ __launch_bounds__(256) void dice_gt_extract(const float4* __restrict__ ann4,
                                                       int* __restrict__ gt_label,
                                                       int* __restrict__ gt_count) {
  const int gid = blockIdx.x * blockDim.x + threadIdx.x;   // one float4 each
  float4 v = ann4[gid];
  const int e = gid << 2;
  float vals[4] = {v.x, v.y, v.z, v.w};
#pragma unroll
  for (int j = 0; j < 4; ++j) {
    if (vals[j] > 0.5f) {                 // one-hot hit (rare: 1 in 512)
      int ee = e + j;                     // ee = ((b*K + k)*HW + hw)
      int hw = ee & (HW - 1);
      int bk = ee >> 14;
      int k  = bk & (KCLS - 1);
      int b  = bk >> 9;
      gt_label[b * HW + hw] = k;
      atomicAdd(&gt_count[k], 1);
    }
  }
}

// ---------------- kernel: WMMA GEMM + argmax (cosine-sim classify) ----------------
// Workgroup = 256 threads (8 waves), owns 256 contiguous pixels (32 px / wave).
// Register tile per wave: 2 M-tiles x 2 N-tiles -> 4 WMMAs per K-chunk,
// 8 ds_load_b128 per K-chunk = 2 LDS loads per WMMA (vs 4 untiled).
// LDS: x-tile 256x512 bf16 (stride 520), emb-tile 32x512 bf16 (stride 520), 512 rsqrt f32.
#define XSTR 520
#define ESTR 520
#define WGPX 256
#define SMEM_BYTES (WGPX * XSTR * 2 + 32 * ESTR * 2 + KCLS * 4)   // 301568 B < 320 KB

__global__ __launch_bounds__(256) void dice_gemm_argmax(const float* __restrict__ outp,
                                                        const __bf16* __restrict__ embb,
                                                        const float* __restrict__ rsb,
                                                        int* __restrict__ pred) {
  extern __shared__ __align__(16) char smem[];
  __bf16* xs = (__bf16*)smem;                                   // [256][XSTR]
  __bf16* es = (__bf16*)(smem + WGPX * XSTR * 2);               // [32][ESTR]
  float*  rs = (float*) (smem + WGPX * XSTR * 2 + 32 * ESTR * 2);

  const int n0  = blockIdx.x * WGPX;      // first pixel of this workgroup
  const int b   = n0 >> 14;               // batch (256 | HW so no straddle)
  const int hw0 = n0 & (HW - 1);
  const float* gx = outp + (size_t)b * (CH * HW) + hw0;

  // Stage x-tile: coalesced float4 along pixels per channel, transpose into LDS as bf16.
  for (int i = threadIdx.x; i < (WGPX * CH) / 4; i += 256) {
    int c  = i >> 6;                 // 64 float4 groups of pixels per channel
    int p4 = (i & 63) << 2;
    float4 v = *(const float4*)(gx + (size_t)c * HW + p4);
    xs[(p4 + 0) * XSTR + c] = (__bf16)v.x;
    xs[(p4 + 1) * XSTR + c] = (__bf16)v.y;
    xs[(p4 + 2) * XSTR + c] = (__bf16)v.z;
    xs[(p4 + 3) * XSTR + c] = (__bf16)v.w;
  }
  for (int i = threadIdx.x; i < KCLS; i += 256) rs[i] = rsb[i];

  const int lane = threadIdx.x & 31;
  const int wv   = threadIdx.x >> 5;
  const int nrow = lane & 15;        // A: pixel row M ; B: class column N
  const int half = lane >> 4;        // K-split half per 16-bit WMMA layout
  const __bf16* xrow0 = xs + (wv * 32 + nrow) * XSTR;        // M-tile 0
  const __bf16* xrow1 = xs + (wv * 32 + 16 + nrow) * XSTR;   // M-tile 1
  const __bf16* erow0 = es + nrow * ESTR;                    // N-tile 0
  const __bf16* erow1 = es + (16 + nrow) * ESTR;             // N-tile 1

  float best[2][8];
  int   bcls[2][8];
#pragma unroll
  for (int mi = 0; mi < 2; ++mi)
#pragma unroll
    for (int r = 0; r < 8; ++r) { best[mi][r] = -3.4e38f; bcls[mi][r] = 0; }

  for (int t = 0; t < KCLS / 32; ++t) {
    __syncthreads();   // all waves done reading previous emb tile
    // Stage emb tile: 32 classes x 512 channels, 16B chunks.
    for (int i = threadIdx.x; i < (32 * CH) / 8; i += 256) {
      int cc = i >> 6;
      int ch = (i & 63) << 3;
      *(bf16x8*)(es + cc * ESTR + ch) =
          *(const bf16x8*)(embb + (((size_t)(t * 32 + cc)) << 9) + ch);
    }
    __syncthreads();

    f32x8 acc00 = {0.f,0.f,0.f,0.f,0.f,0.f,0.f,0.f};
    f32x8 acc01 = acc00, acc10 = acc00, acc11 = acc00;
#pragma unroll 2
    for (int kc = 0; kc < CH; kc += 32) {
      // 16-bit A/B fragment layout: lanes hold K {half*8..+7} and {16+half*8..+7}
      bf16x8 a00 = *(const bf16x8*)(xrow0 + kc + half * 8);
      bf16x8 a01 = *(const bf16x8*)(xrow0 + kc + 16 + half * 8);
      bf16x8 a10 = *(const bf16x8*)(xrow1 + kc + half * 8);
      bf16x8 a11 = *(const bf16x8*)(xrow1 + kc + 16 + half * 8);
      bf16x8 b00 = *(const bf16x8*)(erow0 + kc + half * 8);
      bf16x8 b01 = *(const bf16x8*)(erow0 + kc + 16 + half * 8);
      bf16x8 b10 = *(const bf16x8*)(erow1 + kc + half * 8);
      bf16x8 b11 = *(const bf16x8*)(erow1 + kc + 16 + half * 8);
      bf16x16 av0 = __builtin_shufflevector(a00, a01, 0,1,2,3,4,5,6,7,8,9,10,11,12,13,14,15);
      bf16x16 av1 = __builtin_shufflevector(a10, a11, 0,1,2,3,4,5,6,7,8,9,10,11,12,13,14,15);
      bf16x16 bv0 = __builtin_shufflevector(b00, b01, 0,1,2,3,4,5,6,7,8,9,10,11,12,13,14,15);
      bf16x16 bv1 = __builtin_shufflevector(b10, b11, 0,1,2,3,4,5,6,7,8,9,10,11,12,13,14,15);
      acc00 = __builtin_amdgcn_wmma_f32_16x16x32_bf16(false, av0, false, bv0, (short)0, acc00, false, false);
      acc01 = __builtin_amdgcn_wmma_f32_16x16x32_bf16(false, av0, false, bv1, (short)0, acc01, false, false);
      acc10 = __builtin_amdgcn_wmma_f32_16x16x32_bf16(false, av1, false, bv0, (short)0, acc10, false, false);
      acc11 = __builtin_amdgcn_wmma_f32_16x16x32_bf16(false, av1, false, bv1, (short)0, acc11, false, false);
    }

    const int cls0 = (t << 5) + nrow;        // N-tile 0 class for this lane
    const int cls1 = cls0 + 16;              // N-tile 1 class for this lane
    const float rc0 = rs[cls0];              // fold rsqrt(b_l2); row norm is argmax-invariant
    const float rc1 = rs[cls1];
#pragma unroll
    for (int r = 0; r < 8; ++r) {
      float v00 = acc00[r] * rc0;            // pixel-tile 0
      float v01 = acc01[r] * rc1;
      if (v00 > best[0][r]) { best[0][r] = v00; bcls[0][r] = cls0; }
      if (v01 > best[0][r]) { best[0][r] = v01; bcls[0][r] = cls1; }
      float v10 = acc10[r] * rc0;            // pixel-tile 1
      float v11 = acc11[r] * rc1;
      if (v10 > best[1][r]) { best[1][r] = v10; bcls[1][r] = cls0; }
      if (v11 > best[1][r]) { best[1][r] = v11; bcls[1][r] = cls1; }
    }
  }

  // C layout: VGPR r, lanes 0-15 -> pixel row r, lanes 16-31 -> row r+8.
  // Argmax across the 16 lanes sharing a pixel; tie-break toward lower class.
#pragma unroll
  for (int mi = 0; mi < 2; ++mi) {
    const int pixbase = n0 + wv * 32 + mi * 16 + half * 8;
#pragma unroll
    for (int r = 0; r < 8; ++r) {
      float v = best[mi][r];
      int   c = bcls[mi][r];
#pragma unroll
      for (int m = 8; m >= 1; m >>= 1) {
        float ov = __shfl_xor(v, m, 32);
        int   oc = __shfl_xor(c, m, 32);
        if (ov > v || (ov == v && oc < c)) { v = ov; c = oc; }
      }
      if (nrow == 0) pred[pixbase + r] = c;
    }
  }
}

// ---------------- kernel: pred/intersection histograms ----------------
__global__ __launch_bounds__(256) void dice_hist(const int* __restrict__ pred,
                                                 const int* __restrict__ gt,
                                                 int* __restrict__ pred_count,
                                                 int* __restrict__ inter) {
  __shared__ int sp[KCLS];
  __shared__ int si[KCLS];
  for (int i = threadIdx.x; i < KCLS; i += 256) { sp[i] = 0; si[i] = 0; }
  __syncthreads();
  for (int n = blockIdx.x * blockDim.x + threadIdx.x; n < NPIX; n += gridDim.x * blockDim.x) {
    int p = pred[n];
    int g = gt[n];
    atomicAdd(&sp[p], 1);
    if (p == g) atomicAdd(&si[p], 1);
  }
  __syncthreads();
  for (int i = threadIdx.x; i < KCLS; i += 256) {
    if (sp[i]) atomicAdd(&pred_count[i], sp[i]);
    if (si[i]) atomicAdd(&inter[i], si[i]);
  }
}

// ---------------- kernel: dice score + masked mean ----------------
__global__ __launch_bounds__(512) void dice_finalize(const int* __restrict__ gtc,
                                                     const int* __restrict__ prc,
                                                     const int* __restrict__ itc,
                                                     float* __restrict__ out) {
  __shared__ float red[KCLS];
  const int k = threadIdx.x;
  float g = (float)gtc[k];
  float p = (float)prc[k];
  float I = (float)itc[k];
  float card  = p + g;
  float score = (2.0f * I + 1e-4f) / fmaxf(card + 1e-4f, 1e-7f);
  float loss  = 1.0f - score;
  red[k] = (g > 0.0f) ? loss : 0.0f;
  __syncthreads();
  for (int s = 256; s > 0; s >>= 1) {
    if (k < s) red[k] += red[k + s];
    __syncthreads();
  }
  if (k == 0) out[0] = red[0] / (float)KCLS;
}

// ---------------- launcher ----------------
extern "C" void kernel_launch(void* const* d_in, const int* in_sizes, int n_in,
                              void* d_out, int out_size, void* d_ws, size_t ws_size,
                              hipStream_t stream) {
  const float* outp = (const float*)d_in[0];      // [BS, C, H, W] f32
  const float* ann  = (const float*)d_in[1];      // [BS, K, H, W] f32 one-hot
  const float* emb  = (const float*)d_in[2];      // [K, C] f32
  float* out        = (float*)d_out;

  char* ws = (char*)d_ws;
  __bf16* embb    = (__bf16*)(ws + WS_EMBB);
  float*  rsb     = (float*) (ws + WS_RSB);
  int*    gtc     = (int*)   (ws + WS_GTC);
  int*    prc     = (int*)   (ws + WS_PRC);
  int*    itc     = (int*)   (ws + WS_INT);
  int*    gt_lbl  = (int*)   (ws + WS_GTL);
  int*    pr_lbl  = (int*)   (ws + WS_PRL);

  dice_emb_prep<<<KCLS, 256, 0, stream>>>(emb, embb, rsb);
  dice_zero<<<1, 256, 0, stream>>>(gtc);   // gtc/prc/itc are contiguous (3*512 ints)
  dice_gt_extract<<<(BS * KCLS * HW) / 4 / 256, 256, 0, stream>>>(
      (const float4*)ann, gt_lbl, gtc);
  dice_gemm_argmax<<<NPIX / WGPX, 256, SMEM_BYTES, stream>>>(outp, embb, rsb, pr_lbl);
  dice_hist<<<128, 256, 0, stream>>>(pr_lbl, gt_lbl, prc, itc);
  dice_finalize<<<1, 512, 0, stream>>>(gtc, prc, itc, out);
}
